// dspo_model_9491877724839
// MI455X (gfx1250) — matrix operand order repaired and checked by reading, not compiled, not run on hardware
//
#include <hip/hip_runtime.h>
#include <stdint.h>
#include <stddef.h>

// ---------------------------------------------------------------------------
// Kalman predictor rollout as parallel-prefix of affine maps:
//   M = A - L*H;  x_{k+1} = M x_k + L y_k;  yhat_k = H x_k
// Phase 1 (up-sweep):  per 16-step block b: c_b via 16 batched GEMM rounds
// Phase 2 (scan):      X_{b+1} = P X_b + c_b, P = M^16, Kogge-Stone doubling
// Phase 3 (down-sweep):re-run 16 in-block steps batched, yhat = H*x each step
// Heavy math = bf16 WMMA GEMMs (v_wmma_f32_16x16x32_bf16), fp32 accumulate,
// with double-buffered LDS staged by global_load_async_to_lds_b128 (ASYNCcnt).
// Working set ~40 MB -> L2 resident (192 MB); compute-bound on WMMA.
// ---------------------------------------------------------------------------

typedef __bf16 bf16_t;
typedef __bf16 v16bf __attribute__((ext_vector_type(16)));
typedef float  v8f   __attribute__((ext_vector_type(8)));

#define TM 128
#define TN 256
#define TK 32

struct GemmArgs {
  // D(M x N) = alpha * (A1@B1 [+ A2@B2]) [+ Cin]; strides in elements.
  // A operands are row-major k-contiguous: elem(r,k) = a[r*asr + k].
  // B operands are k-contiguous:           elem(k,c) = b[c*bsc + k].
  const bf16_t* a1; long long a1sr; int k1;
  const bf16_t* b1; long long b1sc;
  const bf16_t* a2; long long a2sr; int k2;   // a2==nullptr -> single term
  const bf16_t* b2; long long b2sc;
  const void*   cin; long long csr, csc; int cinF32; // cin==nullptr -> none
  float alpha;
  bf16_t* dbf;  long long dbsr,  dbsc;   // optional bf16 out
  bf16_t* dbf2; long long db2sr, db2sc;  // optional 2nd bf16 out (transpose)
  float*  df32; long long dfsr,  dfsc;   // optional f32 out
};

__device__ __forceinline__ v16bf ld_frag(const bf16_t* p0, const bf16_t* p1) {
  union { v16bf v; uint4 q[2]; } u;
  u.q[0] = *reinterpret_cast<const uint4*>(p0);
  u.q[1] = *reinterpret_cast<const uint4*>(p1);
  return u.v;
}

// Async 16-byte global->LDS copy (gfx1250). LDS address = low 32 bits of the
// generic pointer (ISA 10.2: LDS aperture uses addr[31:0]).
__device__ __forceinline__ void async_b128(const bf16_t* lds, const bf16_t* src) {
  unsigned ldsOff = (unsigned)(size_t)lds;
  asm volatile("global_load_async_to_lds_b128 %0, %1, off"
               :: "v"(ldsOff), "v"(src) : "memory");
}

__global__ __launch_bounds__(256) void gemm_bf16_wmma(GemmArgs g) {
  __shared__ __align__(16) bf16_t lA[2][TM * TK]; // [buf][row][k]
  __shared__ __align__(16) bf16_t lB[2][TN * TK]; // [buf][col][k]
  const int tid  = threadIdx.x;
  const int lane = tid & 31;
  const int wave = tid >> 5;
  const int wrow = (wave >> 2) * 64;   // 2 wave rows x 64
  const int wcol = (wave & 3) * 64;    // 4 wave cols x 64
  const long long rowBlock = (long long)blockIdx.y * TM;
  const long long colBlock = (long long)blockIdx.x * TN;

  v8f acc[4][4];
#pragma unroll
  for (int mi = 0; mi < 4; ++mi)
#pragma unroll
    for (int ni = 0; ni < 4; ++ni)
#pragma unroll
      for (int v = 0; v < 8; ++v) acc[mi][ni][v] = 0.0f;

  const int nk = (g.k1 + (g.a2 ? g.k2 : 0)) / TK;

  // Issue one K-tile's async stage (6 x b128 per thread: 2 A + 4 B chunks).
  auto stage = [&](int it, int buf) {
    const int kk = it * TK;
    const bf16_t* ap; long long asr;
    const bf16_t* bp; long long bsc;
    int kloc;
    if (kk < g.k1) { ap = g.a1; asr = g.a1sr; bp = g.b1; bsc = g.b1sc; kloc = kk; }
    else           { ap = g.a2; asr = g.a2sr; bp = g.b2; bsc = g.b2sc; kloc = kk - g.k1; }
#pragma unroll
    for (int ci = tid; ci < (TM * TK) / 8; ci += 256) {
      int r = ci >> 2, kc = (ci & 3) * 8;
      async_b128(&lA[buf][r * TK + kc], ap + (rowBlock + r) * asr + (kloc + kc));
    }
#pragma unroll
    for (int ci = tid; ci < (TN * TK) / 8; ci += 256) {
      int c = ci >> 2, kc = (ci & 3) * 8;
      async_b128(&lB[buf][c * TK + kc], bp + (colBlock + c) * bsc + (kloc + kc));
    }
  };

  stage(0, 0);
  for (int it = 0; it < nk; ++it) {
    const int cur = it & 1;
    if (it + 1 < nk) {
      stage(it + 1, cur ^ 1);                       // prefetch next tile
      asm volatile("s_wait_asynccnt 0x6" ::: "memory"); // cur tile's 6 done
    } else {
      asm volatile("s_wait_asynccnt 0x0" ::: "memory");
    }
    __syncthreads();

    // Fragment layouts per CDNA5 ISA 7.12.2 (16-bit A 16x32 / B 32x16):
    //   A: lane row = lane%16; K = {kh..kh+7, 16+kh..16+kh+7}, kh=(lane/16)*8
    //   B: lane col = lane%16; K = {khb..khb+7, 16+khb..16+khb+7}, khb=(lane/16)*8
    const int kha = (lane >> 4) * 8;
    const int khb = (lane >> 4) * 8;
    const int l15 = lane & 15;
    v16bf af[4], bfr[4];
#pragma unroll
    for (int mi = 0; mi < 4; ++mi) {
      const int r = wrow + mi * 16 + l15;
      af[mi] = ld_frag(&lA[cur][r * TK + kha], &lA[cur][r * TK + 16 + kha]);
    }
#pragma unroll
    for (int ni = 0; ni < 4; ++ni) {
      const int c = wcol + ni * 16 + l15;
      bfr[ni] = ld_frag(&lB[cur][c * TK + khb], &lB[cur][c * TK + 16 + khb]);
    }
#pragma unroll
    for (int mi = 0; mi < 4; ++mi)
#pragma unroll
      for (int ni = 0; ni < 4; ++ni)
        acc[mi][ni] = __builtin_amdgcn_wmma_f32_16x16x32_bf16(
            false, af[mi], false, bfr[ni], (short)0, acc[mi][ni], false, false);
    __syncthreads();   // all reads done before next iteration overwrites
  }

  // C/D layout: VGPR v, lanes 0-15 -> row v, lanes 16-31 -> row 8+v; col lane%16
  const int lrow = (lane >> 4) * 8;
  const int lcol = lane & 15;
#pragma unroll
  for (int mi = 0; mi < 4; ++mi)
#pragma unroll
    for (int ni = 0; ni < 4; ++ni) {
      const long long row0 = rowBlock + wrow + mi * 16 + lrow; // 8 consec rows
      const long long col  = colBlock + wcol + ni * 16 + lcol;
      float vals[8];
#pragma unroll
      for (int v = 0; v < 8; ++v) vals[v] = g.alpha * acc[mi][ni][v];
      if (g.cin) {
        if (g.cinF32) {
#pragma unroll
          for (int v = 0; v < 8; ++v)
            vals[v] += ((const float*)g.cin)[(row0 + v) * g.csr + col * g.csc];
        } else if (g.csr == 1) {  // column-major bf16 addend: one b128 load
          union { uint4 q; bf16_t h[8]; } u;
          u.q = *(const uint4*)&((const bf16_t*)g.cin)[row0 + col * g.csc];
#pragma unroll
          for (int v = 0; v < 8; ++v) vals[v] += (float)u.h[v];
        } else {
#pragma unroll
          for (int v = 0; v < 8; ++v)
            vals[v] += (float)((const bf16_t*)g.cin)[(row0 + v) * g.csr + col * g.csc];
        }
      }
      if (g.dbf) {
        if (g.dbsr == 1) {        // column-contiguous: one b128 store
          union { uint4 q; bf16_t h[8]; } u;
#pragma unroll
          for (int v = 0; v < 8; ++v) u.h[v] = (bf16_t)vals[v];
          *(uint4*)&g.dbf[row0 + col * g.dbsc] = u.q;
        } else {
#pragma unroll
          for (int v = 0; v < 8; ++v)
            g.dbf[(row0 + v) * g.dbsr + col * g.dbsc] = (bf16_t)vals[v];
        }
      }
      if (g.dbf2) {               // transposed copy (db2sr == 1 in practice)
        if (g.db2sr == 1) {
          union { uint4 q; bf16_t h[8]; } u;
#pragma unroll
          for (int v = 0; v < 8; ++v) u.h[v] = (bf16_t)vals[v];
          *(uint4*)&g.dbf2[row0 + col * g.db2sc] = u.q;
        } else {
#pragma unroll
          for (int v = 0; v < 8; ++v)
            g.dbf2[(row0 + v) * g.db2sr + col * g.db2sc] = (bf16_t)vals[v];
        }
      }
      if (g.df32) {
#pragma unroll
        for (int v = 0; v < 8; ++v)
          g.df32[(row0 + v) * g.dfsr + col * g.dfsc] = vals[v];
      }
    }
}

// dst[c*rows + r] = (bf16) src[r*cols + c]   (cols = 1<<logCols)
__global__ void conv_t(const float* __restrict__ src, bf16_t* __restrict__ dst,
                       int rows, int logCols) {
  long long idx = (long long)blockIdx.x * blockDim.x + threadIdx.x;
  int r = (int)(idx >> logCols);
  int c = (int)(idx & ((1 << logCols) - 1));
  dst[(long long)c * rows + r] = (bf16_t)src[idx];
}

__global__ void conv_cast(const float* __restrict__ src, bf16_t* __restrict__ dst) {
  long long idx = (long long)blockIdx.x * blockDim.x + threadIdx.x;
  dst[idx] = (bf16_t)src[idx];
}

__global__ void fill_zero(bf16_t* __restrict__ dst) {
  long long idx = (long long)blockIdx.x * blockDim.x + threadIdx.x;
  dst[idx] = (bf16_t)0.0f;
}

static void launch_gemm(hipStream_t s, int Md, int Nd,
    const bf16_t* a1, long long a1sr, int k1,
    const bf16_t* b1, long long b1sc,
    const bf16_t* a2, long long a2sr, int k2,
    const bf16_t* b2, long long b2sc,
    const void* cin, long long csr, long long csc, int cinF32,
    float alpha,
    bf16_t* dbf, long long dbsr, long long dbsc,
    bf16_t* dbf2, long long db2sr, long long db2sc,
    float* df32, long long dfsr, long long dfsc) {
  GemmArgs g;
  g.a1 = a1; g.a1sr = a1sr; g.k1 = k1;
  g.b1 = b1; g.b1sc = b1sc;
  g.a2 = a2; g.a2sr = a2sr; g.k2 = k2;
  g.b2 = b2; g.b2sc = b2sc;
  g.cin = cin; g.csr = csr; g.csc = csc; g.cinF32 = cinF32;
  g.alpha = alpha;
  g.dbf = dbf; g.dbsr = dbsr; g.dbsc = dbsc;
  g.dbf2 = dbf2; g.db2sr = db2sr; g.db2sc = db2sc;
  g.df32 = df32; g.dfsr = dfsr; g.dfsc = dfsc;
  dim3 grid((unsigned)(Nd / TN), (unsigned)(Md / TM), 1);
  gemm_bf16_wmma<<<grid, dim3(256, 1, 1), 0, s>>>(g);
}

static inline char* wsAlloc(char*& p, size_t bytes) {
  char* r = p;
  p += (bytes + 255) & ~(size_t)255;
  return r;
}

extern "C" void kernel_launch(void* const* d_in, const int* in_sizes, int n_in,
                              void* d_out, int out_size, void* d_ws, size_t ws_size,
                              hipStream_t stream) {
  (void)in_sizes; (void)n_in; (void)out_size; (void)ws_size;
  const float* y = (const float*)d_in[0];   // (128, 32768)
  const float* A = (const float*)d_in[1];   // (1024, 1024)
  const float* H = (const float*)d_in[2];   // (128, 1024)
  const float* L = (const float*)d_in[3];   // (1024, 128)
  float* out = (float*)d_out;               // (128, 32768)

  const int n = 1024, m = 128, N = 32768, S = 16, B = N / S; // B = 2048
  const int PAD = 1024;  // guard columns (max scan shift), pre-zeroed

  // Workspace carve-out (~40 MB)
  char* p = (char*)d_ws;
  bf16_t* Ybf  = (bf16_t*)wsAlloc(p, (size_t)N * m * 2);  // y, timestep-major
  bf16_t* Mbf  = (bf16_t*)wsAlloc(p, (size_t)n * n * 2);  // M row-major (A-side)
  bf16_t* MbfT = (bf16_t*)wsAlloc(p, (size_t)n * n * 2);  // M transposed (B-side)
  bf16_t* Lbf  = (bf16_t*)wsAlloc(p, (size_t)n * m * 2);
  bf16_t* Hbf  = (bf16_t*)wsAlloc(p, (size_t)m * n * 2);  // H row-major (A-side)
  bf16_t* HbfT = (bf16_t*)wsAlloc(p, (size_t)m * n * 2);  // H transposed (B-side)
  bf16_t* PA   = (bf16_t*)wsAlloc(p, (size_t)n * n * 2);
  bf16_t* PAT  = (bf16_t*)wsAlloc(p, (size_t)n * n * 2);
  bf16_t* PB   = (bf16_t*)wsAlloc(p, (size_t)n * n * 2);
  bf16_t* PBT  = (bf16_t*)wsAlloc(p, (size_t)n * n * 2);
  bf16_t* R0   = (bf16_t*)wsAlloc(p, (size_t)n * (PAD + B) * 2); // pad + states
  bf16_t* R1   = (bf16_t*)wsAlloc(p, (size_t)n * (PAD + B) * 2);
  bf16_t* R2   = (bf16_t*)wsAlloc(p, (size_t)n * (PAD + B) * 2);
  bf16_t* bufs[3] = {R0 + (size_t)n * PAD, R1 + (size_t)n * PAD, R2 + (size_t)n * PAD};

  // --- conversions + zero-fill (pads and initial state) ---
  conv_t<<<(N * m) / 256, 256, 0, stream>>>(y, Ybf, m, 15);    // Ybf[t*128+i]
  conv_t<<<(m * n) / 256, 256, 0, stream>>>(H, HbfT, m, 10);   // HbfT[c*128+i]
  conv_cast<<<(m * n) / 256, 256, 0, stream>>>(H, Hbf);
  conv_cast<<<(n * m) / 256, 256, 0, stream>>>(L, Lbf);
  fill_zero<<<((size_t)n * (PAD + B)) / 256, 256, 0, stream>>>(R0); // pad+state0
  fill_zero<<<((size_t)n * PAD) / 256, 256, 0, stream>>>(R1);
  fill_zero<<<((size_t)n * PAD) / 256, 256, 0, stream>>>(R2);

  // --- M = A - L@H  (alpha=-1, Cin = A fp32); emit M and M^T ---
  launch_gemm(stream, n, n,
              Lbf, m, m,             // A1 = L (1024x128 row-major)
              HbfT, m,               // B1 = H via transpose (k-contiguous)
              nullptr, 0, 0, nullptr, 0,
              A, n, 1, 1,            // Cin = A fp32 row-major
              -1.0f,
              Mbf, n, 1,             // row-major
              MbfT, 1, n,            // transposed (vector store path)
              nullptr, 0, 0);

  auto square = [&](const bf16_t* src, const bf16_t* srcT,
                    bf16_t* dst, bf16_t* dstT) {
    launch_gemm(stream, n, n,
                src, n, n,
                srcT, n,             // k-contiguous via transpose
                nullptr, 0, 0, nullptr, 0,
                nullptr, 0, 0, 0,
                1.0f, dst, n, 1, dstT, 1, n, nullptr, 0, 0);
  };
  // P = M^16 via 4 squarings -> (PB, PBT)
  square(Mbf, MbfT, PA, PAT); square(PA, PAT, PB, PBT);
  square(PB, PBT, PA, PAT);   square(PA, PAT, PB, PBT);

  // --- Phase 1 (up-sweep): c <- M c + L y_j, j = 0..15, batched over blocks ---
  int cc = 0, cn = 1;
  for (int j = 0; j < S; ++j) {
    launch_gemm(stream, n, B,
                Mbf, n, n,
                bufs[cc], n,                          // states col-major
                Lbf, m, m,
                Ybf + (size_t)j * m, (long long)S * m, // Ystep_j
                nullptr, 0, 0, 0,
                1.0f,
                bufs[cn], 1, n,
                nullptr, 0, 0, nullptr, 0, 0);
    int t = cc; cc = cn; cn = t;
  } // result c in bufs[cc]

  // --- Phase 2 (scan): D <- D + P^{2^t} @ shift(D, 2^t), t = 0..10 ---
  bf16_t* pc = PB; bf16_t* pcT = PBT;
  bf16_t* po = PA; bf16_t* poT = PAT;
  int dc = cc;
  for (int t = 0; t < 11; ++t) {
    int dn = (dc + 1) % 3;
    launch_gemm(stream, n, B,
                pc, n, n,
                bufs[dc] - (size_t)(1 << t) * n, n,   // shifted; pad gives zeros
                nullptr, 0, 0, nullptr, 0,
                bufs[dc], 1, n, 0,                    // + D (bf16, csr==1)
                1.0f,
                bufs[dn], 1, n,
                nullptr, 0, 0, nullptr, 0, 0);
    if (t < 10) {
      square(pc, pcT, po, poT);
      bf16_t* t1 = pc; pc = po; po = t1;
      bf16_t* t2 = pcT; pcT = poT; poT = t2;
    }
    dc = dn;
  }
  const bf16_t* Dfin = bufs[dc];                      // D_b;  X_b = D_{b-1}
  bf16_t* xa = bufs[(dc + 1) % 3];
  bf16_t* xb = bufs[(dc + 2) % 3];

  // --- Phase 3 (down-sweep): yhat_j = H x; x <- M x + L y_j ---
  const bf16_t* xc = Dfin - (size_t)n;                // shift 1 col (pad = zeros)
  for (int j = 0; j < S; ++j) {
    launch_gemm(stream, m, B,
                Hbf, n, n,
                xc, n,
                nullptr, 0, 0, nullptr, 0,
                nullptr, 0, 0, 0,
                1.0f,
                nullptr, 0, 0, nullptr, 0, 0,
                out + j, (long long)N, (long long)S); // out[i*N + b*16 + j]
    if (j < S - 1) {
      launch_gemm(stream, n, B,
                  Mbf, n, n,
                  xc, n,
                  Lbf, m, m,
                  Ybf + (size_t)j * m, (long long)S * m,
                  nullptr, 0, 0, 0,
                  1.0f,
                  xa, 1, n,
                  nullptr, 0, 0, nullptr, 0, 0);
      xc = xa;
      bf16_t* tt = xa; xa = xb; xb = tt;
    }
  }
}
